// Decoder_5059471475306
// MI455X (gfx1250) — matrix operand order repaired and checked by reading, not compiled, not run on hardware
//
#include <hip/hip_runtime.h>

typedef __attribute__((ext_vector_type(16))) _Float16     v16h;
typedef __attribute__((ext_vector_type(8)))  float        v8f;
typedef __attribute__((ext_vector_type(4)))  float        v4f;
typedef __attribute__((ext_vector_type(4)))  _Float16     v4h;
typedef __attribute__((ext_vector_type(4)))  unsigned int v4u;

__device__ __forceinline__ float selu_f(float x) {
    const float scale = 1.0507009873554805f;
    const float alpha = 1.6732632423543772f;
    return scale * (x > 0.0f ? x : alpha * (expf(x) - 1.0f));
}

// ---------------------------------------------------------------------------
// WMMA GEMM: out[M x NCOLS] = concat(x[M x XC], pos[M x 3?]) @ W[K x wcols]
// A and W^T staged in LDS as f16, zero-padded to KP (template, mult. of 32).
// Block = 256 threads = 8 waves, 64 rows x NCOLS cols per block.
// Requires M % 64 == 0 (all call sites satisfy this).
// Each wave owns one 16-col tile (nt) -> B fragments preloaded once,
// reused across the 4 M-tiles. Fragments move as 16B ds_load_b128.
// ---------------------------------------------------------------------------
template<int NCOLS, int KP, bool EPI>
__global__ __launch_bounds__(256)
void gemm_wmma_k(const float* __restrict__ x, int xstride, int XC,
                 const float* __restrict__ pos,
                 const float* __restrict__ W, int wcols,
                 const float* __restrict__ bias,
                 float* __restrict__ out, int ostride,
                 int M, int K)
{
    constexpr int NKT = KP / 32;
    __shared__ __align__(16) _Float16 sA[64 * KP];
    __shared__ __align__(16) _Float16 sB[NCOLS * KP];

    const int tid = threadIdx.x;
    const int blockRow = blockIdx.x * 64;

    // ---- stage W transposed: sB[col*KP + k] = W[k][col], packed 4xf16 ----
    for (int i4 = tid; i4 < (NCOLS * KP) / 4; i4 += 256) {
        const int i = i4 * 4;
        const int col = i / KP, k = i - col * KP;
        v4h h = {};
        if (col < wcols) {
            #pragma unroll
            for (int j = 0; j < 4; ++j) {
                const int kk = k + j;
                h[j] = (_Float16)((kk < K) ? W[(size_t)kk * wcols + col] : 0.0f);
            }
        }
        *(v4h*)&sB[i] = h;
    }
    // ---- stage A rows: sA[r*KP + k] = concat(x,pos), packed 4xf16 ----
    for (int i4 = tid; i4 < (64 * KP) / 4; i4 += 256) {
        const int i = i4 * 4;
        const int r = i / KP, k = i - r * KP;
        const int g = blockRow + r;
        v4h h = {};
        if (k + 3 < XC) {                       // fast: 16B global load
            v4f v = *(const v4f*)&x[(size_t)g * xstride + k];
            h[0] = (_Float16)v[0]; h[1] = (_Float16)v[1];
            h[2] = (_Float16)v[2]; h[3] = (_Float16)v[3];
        } else {
            #pragma unroll
            for (int j = 0; j < 4; ++j) {
                const int kk = k + j;
                float v = 0.0f;
                if (kk < XC)     v = x[(size_t)g * xstride + kk];
                else if (kk < K) v = pos[(size_t)g * 3 + (kk - XC)];
                h[j] = (_Float16)v;
            }
        }
        *(v4h*)&sA[i] = h;
    }
    __syncthreads();

    const int wave = tid >> 5;
    const int lane = tid & 31;
    const int half = lane >> 4;
    const int l16  = lane & 15;

    // Tile ownership: NCOLS==128 -> wave = nt (8 col-tiles), 4 M-tiles each.
    //                 NCOLS==16  -> waves 0..3 take mt=wave, nt=0.
    int nt, mt0, mtN;
    if (NCOLS == 128) { nt = wave; mt0 = 0;    mtN = 4; }
    else              { nt = 0;    mt0 = wave; mtN = (wave < 4) ? wave + 1 : wave; }

    union HU { v4u q[2]; v16h h; };

    // Preload all B fragments for this wave's column tile (invariant over mt).
    // B frag: col = nt*16 + l16; half h holds K = kt*32 + 16h .. +15 (contiguous).
    HU Bf[NKT];
    #pragma unroll
    for (int kt = 0; kt < NKT; ++kt) {
        const v4u* bp = (const v4u*)&sB[(nt * 16 + l16) * KP + kt * 32 + half * 16];
        Bf[kt].q[0] = bp[0];
        Bf[kt].q[1] = bp[1];
    }

    for (int mt = mt0; mt < mtN; ++mt) {
        v8f acc = {};
        const _Float16* arow = &sA[(mt * 16 + l16) * KP + half * 16];
        #pragma unroll
        for (int kt = 0; kt < NKT; ++kt) {
            // A frag: row = l16; half h holds K = 16h+{0..15} (contiguous)
            HU Af;
            const v4u* ap = (const v4u*)(arow + kt * 32);
            Af.q[0] = ap[0];
            Af.q[1] = ap[1];
            acc = __builtin_amdgcn_wmma_f32_16x16x32_f16(
                false, Af.h, false, Bf[kt].h, (short)0, acc, false, false);
        }
        // D layout: VGPR r -> row r + 8*half; col = l16.  M % 64 == 0 -> no guard.
        const int gc = nt * 16 + l16;
        const int gr0 = blockRow + mt * 16 + half * 8;
        #pragma unroll
        for (int r = 0; r < 8; ++r) {
            float v = acc[r];
            if (EPI) v = selu_f(v + bias[gc]);
            out[(size_t)(gr0 + r) * ostride + gc] = v;
        }
    }
}

// ---------------------------------------------------------------------------
// ONERA coordinate transform
// ---------------------------------------------------------------------------
__global__ void transform_k(const float* __restrict__ p, float* __restrict__ tp, int N)
{
    int t = blockIdx.x * 256 + threadIdx.x;
    if (t >= N) return;
    float x = p[t * 3], y = p[t * 3 + 1], z = p[t * 3 + 2];
    float nx = x - 0.5773502691896258f * y;                 // tan(30deg)
    float s  = 1.0f + 0.7857142857142857f * (y / 1.1963f);  // (1/0.56 - 1)
    tp[t * 3]     = nx * s;
    tp[t * 3 + 1] = y * s;
    tp[t * 3 + 2] = z * s;
}

// ---------------------------------------------------------------------------
// Brute-force 3-NN, database tiled through LDS (1024 pts = 12KB per tile)
// ---------------------------------------------------------------------------
#define KTILE 1024
__global__ __launch_bounds__(256)
void knn3_k(const float* __restrict__ q, int Nq,
            const float* __restrict__ db, int Nx,
            int* __restrict__ oidx, float* __restrict__ od2)
{
    __shared__ float s[KTILE * 3];
    int t = blockIdx.x * 256 + threadIdx.x;
    float qx = 0.f, qy = 0.f, qz = 0.f;
    if (t < Nq) { qx = q[t * 3]; qy = q[t * 3 + 1]; qz = q[t * 3 + 2]; }
    float b0 = 1e30f, b1 = 1e30f, b2 = 1e30f;
    int   i0 = 0, i1 = 0, i2 = 0;

    for (int base = 0; base < Nx; base += KTILE) {
        __syncthreads();
        for (int i = threadIdx.x; i < KTILE * 3; i += 256) {
            int p = base + i / 3;
            s[i] = (p < Nx) ? db[(size_t)base * 3 + i] : 1e18f;
        }
        __syncthreads();
        const int lim = min(KTILE, Nx - base);
        for (int j = 0; j < lim; ++j) {
            float dx = qx - s[j * 3], dy = qy - s[j * 3 + 1], dz = qz - s[j * 3 + 2];
            float d = dx * dx + dy * dy + dz * dz;
            if (d < b2) {
                int gi = base + j;
                if (d < b0)      { b2 = b1; i2 = i1; b1 = b0; i1 = i0; b0 = d; i0 = gi; }
                else if (d < b1) { b2 = b1; i2 = i1; b1 = d;  i1 = gi; }
                else             { b2 = d;  i2 = gi; }
            }
        }
    }
    if (t < Nq) {
        oidx[t * 3] = i0; oidx[t * 3 + 1] = i1; oidx[t * 3 + 2] = i2;
        od2[t * 3] = b0;  od2[t * 3 + 1] = b1;  od2[t * 3 + 2] = b2;
    }
}

// ---------------------------------------------------------------------------
// knn_interpolate: inverse-squared-distance weighted gather, float4 wide
// C must be a multiple of 4 (used with C = 128).
// ---------------------------------------------------------------------------
__global__ void interp_k(const float* __restrict__ src, const int* __restrict__ idx,
                         const float* __restrict__ d2, float* __restrict__ dst,
                         int Ny, int C)
{
    const int C4 = C >> 2;
    int t = blockIdx.x * 256 + threadIdx.x;
    if (t >= Ny * C4) return;
    int n = t / C4, c = (t - n * C4) * 4;
    float w0 = 1.0f / fmaxf(d2[n * 3],     1e-16f);
    float w1 = 1.0f / fmaxf(d2[n * 3 + 1], 1e-16f);
    float w2 = 1.0f / fmaxf(d2[n * 3 + 2], 1e-16f);
    v4f a = *(const v4f*)&src[(size_t)idx[n * 3]     * C + c];
    v4f b = *(const v4f*)&src[(size_t)idx[n * 3 + 1] * C + c];
    v4f g = *(const v4f*)&src[(size_t)idx[n * 3 + 2] * C + c];
    float inv = 1.0f / (w0 + w1 + w2);
    v4f r = (w0 * a + w1 * b + w2 * g) * inv;
    *(v4f*)&dst[(size_t)n * C + c] = r;
}

// ---------------------------------------------------------------------------
// GCN helpers
// ---------------------------------------------------------------------------
__global__ void deg_init_k(float* deg, int N)
{
    int t = blockIdx.x * 256 + threadIdx.x;
    if (t < N) deg[t] = 1.0f;   // self loop
}

__global__ void deg_edges_k(const int* __restrict__ ei, int E, float* deg)
{
    int t = blockIdx.x * 256 + threadIdx.x;
    if (t < E) atomicAdd(&deg[ei[E + t]], 1.0f);   // col = edge_index[1]
}

__global__ void deg_rsqrt_k(float* deg, int N)
{
    int t = blockIdx.x * 256 + threadIdx.x;
    if (t < N) deg[t] = rsqrtf(deg[t]);
}

// dst = xw * dinv^2 + bias   (self-loop term + bias, before edge scatter)
__global__ void preinit_k(const float* __restrict__ xw, int xws,
                          const float* __restrict__ dinv,
                          const float* __restrict__ bias,
                          float* __restrict__ dst, int dsts, int N, int C)
{
    int t = blockIdx.x * 256 + threadIdx.x;
    if (t >= N * C) return;
    int n = t / C, c = t - n * C;
    float di = dinv[n];
    dst[(size_t)n * dsts + c] = xw[(size_t)n * xws + c] * di * di + bias[c];
}

__global__ void scatter_k(const int* __restrict__ ei, int E,
                          const float* __restrict__ xw, int xws,
                          const float* __restrict__ dinv,
                          float* __restrict__ dst, int dsts, int C)
{
    int t = blockIdx.x * 256 + threadIdx.x;
    if (t >= E * C) return;
    int e = t / C, c = t - e * C;
    int row = ei[e], col = ei[E + e];
    float norm = dinv[row] * dinv[col];
    atomicAdd(&dst[(size_t)col * dsts + c], xw[(size_t)row * xws + c] * norm);
}

__global__ void selu4_k(float* __restrict__ x, int n4)
{
    int t = blockIdx.x * 256 + threadIdx.x;
    if (t >= n4) return;
    v4f v = *(v4f*)&x[t * 4];
    v[0] = selu_f(v[0]); v[1] = selu_f(v[1]);
    v[2] = selu_f(v[2]); v[3] = selu_f(v[3]);
    *(v4f*)&x[t * 4] = v;
}

// ---------------------------------------------------------------------------
// Orchestration
// ---------------------------------------------------------------------------
static inline dim3 g1(long n) { return dim3((unsigned)((n + 255) / 256)); }

extern "C" void kernel_launch(void* const* d_in, const int* in_sizes, int n_in,
                              void* d_out, int out_size, void* d_ws, size_t ws_size,
                              hipStream_t stream)
{
    const float* latent = (const float*)d_in[0];
    const float* pos0   = (const float*)d_in[1];
    const float* pos1   = (const float*)d_in[2];
    const float* pos2   = (const float*)d_in[3];
    const float* Wlin = (const float*)d_in[4];  const float* blin = (const float*)d_in[5];
    const float* W0   = (const float*)d_in[6];  const float* b0   = (const float*)d_in[7];
    const float* W1   = (const float*)d_in[8];  const float* b1   = (const float*)d_in[9];
    const float* W2   = (const float*)d_in[10]; const float* b2   = (const float*)d_in[11];
    const float* W3   = (const float*)d_in[12]; const float* b3   = (const float*)d_in[13];
    const float* W4   = (const float*)d_in[14]; const float* b4   = (const float*)d_in[15];
    const int* ei0 = (const int*)d_in[16];
    const int* ei1 = (const int*)d_in[17];
    const int* ei2 = (const int*)d_in[18];
    float* out = (float*)d_out;

    const int N0 = 16000, N1 = 32000, N2 = 96000;
    const int E0 = 128000, E1 = 256000, E2 = 768000;
    const int HC = 128, IC = 131;

    char* wsp = (char*)d_ws;
    size_t off = 0;
    auto alloc = [&](size_t bytes) -> void* {
        void* p = wsp + off;
        off += (bytes + 255) & ~(size_t)255;
        return p;
    };
    float* B    = (float*)alloc((size_t)N2 * HC * 4);  // ping
    float* C    = (float*)alloc((size_t)N2 * HC * 4);  // pong
    float* TP0  = (float*)alloc((size_t)N0 * 3 * 4);
    float* TP1  = (float*)alloc((size_t)N1 * 3 * 4);
    float* TP2  = (float*)alloc((size_t)N2 * 3 * 4);
    float* DINV = (float*)alloc((size_t)N2 * 4);
    int*   IDX  = (int*)  alloc((size_t)N2 * 3 * 4);
    float* D2   = (float*)alloc((size_t)N2 * 3 * 4);
    if (off > ws_size) return;   // not enough scratch; bail

    // --- transformed coordinates for all levels ---
    transform_k<<<g1(N0), 256, 0, stream>>>(pos0, TP0, N0);
    transform_k<<<g1(N1), 256, 0, stream>>>(pos1, TP1, N1);
    transform_k<<<g1(N2), 256, 0, stream>>>(pos2, TP2, N2);

    // --- x = selu(latent @ W_lin + b_lin)  -> C ---
    gemm_wmma_k<128, 64, true><<<N0 / 64, 256, 0, stream>>>(
        latent, 64, 64, nullptr, Wlin, HC, blin, C, HC, N0, 64);

    // --- dinv for level 0 (shared by both convs) ---
    deg_init_k <<<g1(N0), 256, 0, stream>>>(DINV, N0);
    deg_edges_k<<<g1(E0), 256, 0, stream>>>(ei0, E0, DINV);
    deg_rsqrt_k<<<g1(N0), 256, 0, stream>>>(DINV, N0);

    // --- conv0a: C -> B (gemm), C = accum, selu ---
    gemm_wmma_k<128, 160, false><<<N0 / 64, 256, 0, stream>>>(
        C, HC, HC, pos0, W0, HC, nullptr, B, HC, N0, IC);
    preinit_k<<<g1((long)N0 * HC), 256, 0, stream>>>(B, HC, DINV, b0, C, HC, N0, HC);
    scatter_k<<<g1((long)E0 * HC), 256, 0, stream>>>(ei0, E0, B, HC, DINV, C, HC, HC);
    selu4_k<<<g1((long)N0 * HC / 4), 256, 0, stream>>>(C, N0 * HC / 4);

    // --- conv0b ---
    gemm_wmma_k<128, 160, false><<<N0 / 64, 256, 0, stream>>>(
        C, HC, HC, pos0, W1, HC, nullptr, B, HC, N0, IC);
    preinit_k<<<g1((long)N0 * HC), 256, 0, stream>>>(B, HC, DINV, b1, C, HC, N0, HC);
    scatter_k<<<g1((long)E0 * HC), 256, 0, stream>>>(ei0, E0, B, HC, DINV, C, HC, HC);
    selu4_k<<<g1((long)N0 * HC / 4), 256, 0, stream>>>(C, N0 * HC / 4);

    // --- interp level 0 -> 1: features C(N0) -> B(N1) ---
    knn3_k<<<N1 / 256, 256, 0, stream>>>(TP1, N1, TP0, N0, IDX, D2);
    interp_k<<<g1((long)N1 * HC / 4), 256, 0, stream>>>(C, IDX, D2, B, N1, HC);

    // --- conv1: features in B; C = xw ---
    deg_init_k <<<g1(N1), 256, 0, stream>>>(DINV, N1);
    deg_edges_k<<<g1(E1), 256, 0, stream>>>(ei1, E1, DINV);
    deg_rsqrt_k<<<g1(N1), 256, 0, stream>>>(DINV, N1);
    gemm_wmma_k<128, 160, false><<<N1 / 64, 256, 0, stream>>>(
        B, HC, HC, pos1, W2, HC, nullptr, C, HC, N1, IC);
    preinit_k<<<g1((long)N1 * HC), 256, 0, stream>>>(C, HC, DINV, b2, B, HC, N1, HC);
    scatter_k<<<g1((long)E1 * HC), 256, 0, stream>>>(ei1, E1, C, HC, DINV, B, HC, HC);
    selu4_k<<<g1((long)N1 * HC / 4), 256, 0, stream>>>(B, N1 * HC / 4);

    // --- interp level 1 -> 2: features B(N1) -> C(N2) ---
    knn3_k<<<N2 / 256, 256, 0, stream>>>(TP2, N2, TP1, N1, IDX, D2);
    interp_k<<<g1((long)N2 * HC / 4), 256, 0, stream>>>(B, IDX, D2, C, N2, HC);

    // --- conv2: features in C; B = xw ---
    deg_init_k <<<g1(N2), 256, 0, stream>>>(DINV, N2);
    deg_edges_k<<<g1(E2), 256, 0, stream>>>(ei2, E2, DINV);
    deg_rsqrt_k<<<g1(N2), 256, 0, stream>>>(DINV, N2);
    gemm_wmma_k<128, 160, false><<<N2 / 64, 256, 0, stream>>>(
        C, HC, HC, pos2, W3, HC, nullptr, B, HC, N2, IC);
    preinit_k<<<g1((long)N2 * HC), 256, 0, stream>>>(B, HC, DINV, b3, C, HC, N2, HC);
    scatter_k<<<g1((long)E2 * HC), 256, 0, stream>>>(ei2, E2, B, HC, DINV, C, HC, HC);
    selu4_k<<<g1((long)N2 * HC / 4), 256, 0, stream>>>(C, N2 * HC / 4);

    // --- output conv (5 cols, padded to 16 in the xw buffer; no activation) ---
    gemm_wmma_k<16, 160, false><<<N2 / 64, 256, 0, stream>>>(
        C, HC, HC, pos2, W4, 5, nullptr, B, 16, N2, IC);
    preinit_k<<<g1((long)N2 * 5), 256, 0, stream>>>(B, 16, DINV, b4, out, 5, N2, 5);
    scatter_k<<<g1((long)E2 * 5), 256, 0, stream>>>(ei2, E2, B, 16, DINV, out, 5, 5);
}
